// LatentSpace_DNN_LocallyConnected_Dense_72868415144018
// MI455X (gfx1250) — compile-verified, matrix operand order checked by reading, not run on hardware
//
#include <hip/hip_runtime.h>

typedef __bf16 bf16_t;
typedef __attribute__((ext_vector_type(16))) __bf16 v16bf;
typedef __attribute__((ext_vector_type(8)))  __bf16 v8bf;
typedef __attribute__((ext_vector_type(8)))  float  v8f;
typedef __attribute__((ext_vector_type(4)))  float  v4f;

#define SPLIT  32
#define DDIM   256
#define NUNITS 256
#define BROWS  4096
#define AROW   40   // 32 bf16 payload + 8 pad (16B) -> bank spread, 16B aligned

// ---------------------------------------------------------------------------
// Prep: W[s][d][u] (f32) -> Wt_hi/Wt_lo [s][u][d] (bf16 hi/lo split).
// LDS-tiled 32x32 transpose: coalesced global reads AND writes.
// ---------------------------------------------------------------------------
__global__ __launch_bounds__(256) void prep_w(const float* __restrict__ W,
                                              bf16_t* __restrict__ whi,
                                              bf16_t* __restrict__ wlo) {
    __shared__ float tile[32][33];
    const int tileId = blockIdx.x;               // 32 splits * 64 tiles
    const int s  = tileId >> 6;
    const int t  = tileId & 63;
    const int d0 = (t >> 3) << 5;
    const int u0 = (t & 7)  << 5;
    const int r  = threadIdx.x >> 5;
    const int c  = threadIdx.x & 31;

    const float* Ws = W + (size_t)s * DDIM * NUNITS;
#pragma unroll
    for (int i = 0; i < 4; ++i) {
        int rr = r + i * 8;
        tile[rr][c] = Ws[(size_t)(d0 + rr) * NUNITS + (u0 + c)];
    }
    __syncthreads();

    bf16_t* hi = whi + (size_t)s * NUNITS * DDIM;
    bf16_t* lo = wlo + (size_t)s * NUNITS * DDIM;
#pragma unroll
    for (int i = 0; i < 4; ++i) {
        int rr = r + i * 8;
        float  v = tile[c][rr];                  // = W[s][d0+c][u0+rr]
        bf16_t h = (bf16_t)v;
        bf16_t l = (bf16_t)(v - (float)h);
        size_t idx = (size_t)(u0 + rr) * DDIM + (d0 + c);
        hi[idx] = h;
        lo[idx] = l;
    }
}

// B fragment from global (pre-converted bf16, contiguous per lane).
// CDNA5 16-bit 16x32 layout: lane<16 -> K {0..7, 16..23}, lane>=16 -> {8..15, 24..31}
__device__ __forceinline__ v16bf load_bfrag(const bf16_t* __restrict__ base,
                                            int k, int half) {
    v8bf r0 = *(const v8bf*)(base + k + half * 8);
    v8bf r1 = *(const v8bf*)(base + k + 16 + half * 8);
    return __builtin_shufflevector(r0, r1, 0, 1, 2, 3, 4, 5, 6, 7,
                                           8, 9, 10, 11, 12, 13, 14, 15);
}

// A fragment from the LDS panel (rowbase -> this lane's row, local K 0..31)
__device__ __forceinline__ v16bf ldfrag(const bf16_t* rowbase, int half) {
    v8bf r0 = *(const v8bf*)(rowbase + half * 8);
    v8bf r1 = *(const v8bf*)(rowbase + 16 + half * 8);
    return __builtin_shufflevector(r0, r1, 0, 1, 2, 3, 4, 5, 6, 7,
                                           8, 9, 10, 11, 12, 13, 14, 15);
}

// ---------------------------------------------------------------------------
// Main GEMM: grid (B/64, SPLIT). 256 threads = 8 wave32, waves 2(M) x 4(N).
// A staged in double-buffered LDS (converted once per WG). B fragments
// software-pipelined one step ahead so every load hides under 6 WMMAs.
// Streaming x/out use non-temporal policy to keep W resident in L2.
// ---------------------------------------------------------------------------
__global__ __launch_bounds__(256) void lc_gemm(const float*  __restrict__ x,
                                               const bf16_t* __restrict__ whi,
                                               const bf16_t* __restrict__ wlo,
                                               const float*  __restrict__ bias,
                                               float*        __restrict__ out) {
    __shared__ bf16_t sAhi[2][64 * AROW];        // 2 x 5 KB
    __shared__ bf16_t sAlo[2][64 * AROW];        // 2 x 5 KB

    const int s     = blockIdx.y;
    const int mWG   = blockIdx.x * 64;
    const int tid   = threadIdx.x;
    const int wave  = tid >> 5;
    const int lane  = tid & 31;
    const int waveM = wave >> 2;                 // 0..1
    const int waveN = wave & 3;                  // 0..3
    const int half  = lane >> 4;
    const int l16   = lane & 15;
    const int nBase = waveN * 64;

    // ---- fill role: thread loads 8 consecutive x floats per k-step --------
    const int frow = tid >> 2;                   // 0..63
    const int fq   = tid & 3;                    // 0..3
    const float* xf = x + ((size_t)(mWG + frow) * SPLIT + s) * DDIM + fq * 8;
    const int foff = frow * AROW + fq * 8;

    auto fill = [&](int buf, int k) {
        const v4f* p = (const v4f*)(xf + k);
        v4f f0 = __builtin_nontemporal_load(p);      // stream: don't pollute L2
        v4f f1 = __builtin_nontemporal_load(p + 1);
        __builtin_prefetch(xf + k + 64, 0, 1);       // two k-steps ahead
        float f[8] = {f0.x, f0.y, f0.z, f0.w, f1.x, f1.y, f1.z, f1.w};
        v8bf h8, l8;
#pragma unroll
        for (int e = 0; e < 8; ++e) {
            bf16_t h = (bf16_t)f[e];
            h8[e] = h;
            l8[e] = (bf16_t)(f[e] - (float)h);
        }
        *(v8bf*)&sAhi[buf][foff] = h8;
        *(v8bf*)&sAlo[buf][foff] = l8;
    };

    // ---- compute role: per-lane LDS row bases for the two m-tiles ---------
    const int r0 = (waveM * 32 + l16) * AROW;
    const int r1 = r0 + 16 * AROW;

    // B column base pointers (Wt layout [s][u][d], bf16) — L2-resident
    const bf16_t* bh[4];
    const bf16_t* bl[4];
#pragma unroll
    for (int j = 0; j < 4; ++j) {
        size_t off = ((size_t)s * NUNITS + (nBase + j * 16 + l16)) * DDIM;
        bh[j] = whi + off;
        bl[j] = wlo + off;
    }

    v8f acc[2][4];
#pragma unroll
    for (int t = 0; t < 2; ++t)
#pragma unroll
        for (int j = 0; j < 4; ++j)
            acc[t][j] = (v8f){0.f, 0.f, 0.f, 0.f, 0.f, 0.f, 0.f, 0.f};

    fill(0, 0);
    // Pre-load first B fragment pair (j=0, k=0)
    v16bf bhv = load_bfrag(bh[0], 0, half);
    v16bf blv = load_bfrag(bl[0], 0, half);
    __syncthreads();

#pragma unroll
    for (int ks = 0; ks < 8; ++ks) {
        const int buf = ks & 1;
        const int k   = ks * 32;

        if (ks < 7) fill(buf ^ 1, k + 32);       // overlap next A panel

        v16bf a0h = ldfrag(&sAhi[buf][r0], half);
        v16bf a0l = ldfrag(&sAlo[buf][r0], half);
        v16bf a1h = ldfrag(&sAhi[buf][r1], half);
        v16bf a1l = ldfrag(&sAlo[buf][r1], half);

#pragma unroll
        for (int j = 0; j < 4; ++j) {
            // Issue next fragment's loads BEFORE this j's WMMAs (1-step SW pipe)
            const bool haveNext = (j < 3) || (ks < 7);
            const int  jn = (j + 1) & 3;
            const int  kn = (j < 3) ? k : k + 32;
            v16bf bhn = bhv, bln = blv;
            if (haveNext) {
                bhn = load_bfrag(bh[jn], kn, half);
                bln = load_bfrag(bl[jn], kn, half);
            }

            acc[0][j] = __builtin_amdgcn_wmma_f32_16x16x32_bf16(
                false, a0h, false, bhv, (short)0, acc[0][j], false, false);
            acc[1][j] = __builtin_amdgcn_wmma_f32_16x16x32_bf16(
                false, a1h, false, bhv, (short)0, acc[1][j], false, false);
            acc[0][j] = __builtin_amdgcn_wmma_f32_16x16x32_bf16(
                false, a0h, false, blv, (short)0, acc[0][j], false, false);
            acc[1][j] = __builtin_amdgcn_wmma_f32_16x16x32_bf16(
                false, a1h, false, blv, (short)0, acc[1][j], false, false);
            acc[0][j] = __builtin_amdgcn_wmma_f32_16x16x32_bf16(
                false, a0l, false, bhv, (short)0, acc[0][j], false, false);
            acc[1][j] = __builtin_amdgcn_wmma_f32_16x16x32_bf16(
                false, a1l, false, bhv, (short)0, acc[1][j], false, false);

            bhv = bhn;
            blv = bln;
        }

        __syncthreads();                         // panel consumed / next ready
    }

    // Epilogue: +bias, ReLU, NT store. C/D: lane = col, VGPR r -> M = r+8*half
#pragma unroll
    for (int j = 0; j < 4; ++j) {
        const int col = nBase + j * 16 + l16;
        const float bv = bias[s * NUNITS + col];
#pragma unroll
        for (int t = 0; t < 2; ++t) {
            const int rbase = mWG + waveM * 32 + t * 16 + half * 8;
#pragma unroll
            for (int r = 0; r < 8; ++r) {
                float v = acc[t][j][r] + bv;
                v = v > 0.f ? v : 0.f;
                __builtin_nontemporal_store(
                    v, &out[((size_t)(rbase + r) * SPLIT + s) * DDIM + col]);
            }
        }
    }
}

extern "C" void kernel_launch(void* const* d_in, const int* in_sizes, int n_in,
                              void* d_out, int out_size, void* d_ws, size_t ws_size,
                              hipStream_t stream) {
    (void)in_sizes; (void)n_in; (void)out_size; (void)ws_size;
    const float* x = (const float*)d_in[0];
    const float* W = (const float*)d_in[1];
    const float* b = (const float*)d_in[2];
    float* out = (float*)d_out;

    bf16_t* whi = (bf16_t*)d_ws;                               // 4 MB
    bf16_t* wlo = whi + (size_t)SPLIT * DDIM * NUNITS;         // 4 MB

    prep_w<<<dim3(SPLIT * 64), dim3(256), 0, stream>>>(W, whi, wlo);
    lc_gemm<<<dim3(BROWS / 64, SPLIT), dim3(256), 0, stream>>>(x, whi, wlo, b, out);
}
